// LSSFPN_24592982737578
// MI455X (gfx1250) — matrix operand order repaired and checked by reading, not compiled, not run on hardware
//
#include <hip/hip_runtime.h>
#include <hip/hip_bf16.h>

// ---------------------------------------------------------------------------
// LSS / BEVDepth view transformer for MI455X (gfx1250, wave32, WMMA).
// Heavy work = two 3x3 512->512 convs as implicit GEMM on v_wmma_f32_16x16x32_f16.
// Activations are channels-last f16 with a 1-pixel zero halo, so the conv
// K-loop has no bounds checks: pure global_load_b128 + v_wmma.
// Each wave computes a 32(co) x 32(pos) tile = 4 accumulators, reusing each
// A/B fragment twice (halves load traffic vs a 16x16 wave tile).
// ---------------------------------------------------------------------------

typedef _Float16 v8h  __attribute__((ext_vector_type(8)));
typedef _Float16 v16h __attribute__((ext_vector_type(16)));
typedef float    v8f  __attribute__((ext_vector_type(8)));

#define CI    512
#define CMID  512
#define COUT_TOT 192   // 112 depth + 80 context
#define NDEPTH 112
#define NCTX   80
#define NIMG   12      // B*N
#define FH    16
#define FW    44
#define NPOS  704      // FH*FW
#define PFH   18       // padded
#define PFW   46
#define PPOS  828      // PFH*PFW
#define NX    128
#define NY    128

// ---------------------------------------------------------------------------
// Fragment loaders (contiguous k order chosen by this code, wave32):
// A 16x32 f16 : lane (0-15 -> M=lane, 16-31 -> M=lane-16), half=lane>>4
//   elements 0..7  = K[kbase + half*16     .. +7]
//   elements 8..15 = K[kbase + half*16 + 8 .. +7]
// B 32x16 f16 : lane half h holds 16 consecutive K values (K = kbase + h*16 + j)
// ---------------------------------------------------------------------------
__device__ inline v16h load_fragA(const _Float16* __restrict__ p) {
    v8h lo = *reinterpret_cast<const v8h*>(p);
    v8h hi = *reinterpret_cast<const v8h*>(p + 8);
    v16h r;
#pragma unroll
    for (int j = 0; j < 8; ++j) { r[j] = lo[j]; r[8 + j] = hi[j]; }
    return r;
}
__device__ inline v16h load_fragB(const _Float16* __restrict__ p) {
    v8h lo = *reinterpret_cast<const v8h*>(p);
    v8h hi = *reinterpret_cast<const v8h*>(p + 8);
    v16h r;
#pragma unroll
    for (int j = 0; j < 8; ++j) { r[j] = lo[j]; r[8 + j] = hi[j]; }
    return r;
}
#define WMMA_F16(A, B, C) \
    __builtin_amdgcn_wmma_f32_16x16x32_f16(false, (A), false, (B), (short)0, (C), false, false)

// ---------------------------------------------------------------------------
// 3x3 conv as implicit GEMM + fused BN(scale/bias) + ReLU.
// X: [NIMG][PPOS][CI] f16 padded channels-last, W: [9][CO][CI] f16,
// Y: [NIMG][PPOS][CO] f16 (interior written; halo pre-zeroed).
// block = 256 threads = 8 waves; wave tile 32(co) x 32(pos);
// block tile 128(co) x 64(pos). grid = (704/64, 512/128, 12) = (11, 4, 12).
// ---------------------------------------------------------------------------
__global__ __launch_bounds__(256) void lss_conv3x3_wmma(
    const _Float16* __restrict__ X, const _Float16* __restrict__ W,
    const float* __restrict__ scale, const float* __restrict__ bias,
    _Float16* __restrict__ Y)
{
    const int img   = blockIdx.z;
    const int wave  = threadIdx.x >> 5;
    const int lane  = threadIdx.x & 31;
    const int khalf = lane >> 4;
    const int mn    = lane & 15;

    const int coBase  = blockIdx.y * 128 + (wave & 3) * 32;
    const int posBase = blockIdx.x * 64 + (wave >> 2) * 32;
    const int pos0 = posBase + mn;
    const int pos1 = pos0 + 16;
    const int h0 = pos0 / FW, w0 = pos0 % FW;
    const int h1 = pos1 / FW, w1 = pos1 % FW;

    // padded interior pixel index for tap (dy,dx): (h+1+dy)*PFW + (w+1+dx)
    const _Float16* ximg = X + (size_t)img * PPOS * CI;

    v8f acc00 = {}, acc01 = {}, acc10 = {}, acc11 = {};
#pragma unroll 1
    for (int t = 0; t < 9; ++t) {
        const int dy = t / 3 - 1, dx = t % 3 - 1;
        const _Float16* aptr0 = W + ((size_t)(t * CMID + coBase + mn)) * CI + khalf * 16;
        const _Float16* aptr1 = aptr0 + (size_t)16 * CI;
        const _Float16* bptr0 = ximg + ((size_t)((h0 + 1 + dy) * PFW + (w0 + 1 + dx))) * CI + khalf * 16;
        const _Float16* bptr1 = ximg + ((size_t)((h1 + 1 + dy) * PFW + (w1 + 1 + dx))) * CI + khalf * 16;
#pragma unroll 2
        for (int kk = 0; kk < CI; kk += 32) {
            v16h a0 = load_fragA(aptr0 + kk);
            v16h a1 = load_fragA(aptr1 + kk);
            v16h b0 = load_fragB(bptr0 + kk);
            v16h b1 = load_fragB(bptr1 + kk);
            acc00 = WMMA_F16(a0, b0, acc00);
            acc01 = WMMA_F16(a0, b1, acc01);
            acc10 = WMMA_F16(a1, b0, acc10);
            acc11 = WMMA_F16(a1, b1, acc11);
        }
    }
    // C/D layout: element r -> M = r + khalf*8, N = mn -> 8 consecutive co.
#pragma unroll
    for (int fa = 0; fa < 2; ++fa) {
        const int co0 = coBase + fa * 16 + khalf * 8;
#pragma unroll
        for (int fb = 0; fb < 2; ++fb) {
            const v8f acc = fa == 0 ? (fb == 0 ? acc00 : acc01)
                                    : (fb == 0 ? acc10 : acc11);
            const int hh = (fb == 0 ? h0 : h1), ww = (fb == 0 ? w0 : w1);
            v8h ov;
#pragma unroll
            for (int r = 0; r < 8; ++r) {
                float v = acc[r] * scale[co0 + r] + bias[co0 + r];
                ov[r] = (_Float16)(v > 0.f ? v : 0.f);
            }
            *reinterpret_cast<v8h*>(
                Y + ((size_t)img * PPOS + (hh + 1) * PFW + (ww + 1)) * CMID + co0) = ov;
        }
    }
}

// ---------------------------------------------------------------------------
// 1x1 conv (512 -> 192) + bias; reads padded X, writes unpadded f32
// HF [NIMG][NPOS][192]. 16x16 wave tile. grid = (22, 3, 12).
// ---------------------------------------------------------------------------
__global__ __launch_bounds__(256) void lss_conv1x1_wmma(
    const _Float16* __restrict__ X, const _Float16* __restrict__ W,
    const float* __restrict__ bias, float* __restrict__ Y)
{
    const int img   = blockIdx.z;
    const int wave  = threadIdx.x >> 5;
    const int lane  = threadIdx.x & 31;
    const int khalf = lane >> 4;
    const int mn    = lane & 15;

    const int coTile = blockIdx.y * 64 + (wave & 3) * 16;
    const int pos    = blockIdx.x * 32 + (wave >> 2) * 16 + mn;
    const int h = pos / FW, w = pos % FW;

    const _Float16* aptr = W + ((size_t)(coTile + mn)) * CI + khalf * 16;
    const _Float16* bptr = X + ((size_t)img * PPOS + (h + 1) * PFW + (w + 1)) * CI + khalf * 16;

    v8f acc = {};
#pragma unroll 4
    for (int kk = 0; kk < CI; kk += 32) {
        v16h a = load_fragA(aptr + kk);
        v16h b = load_fragB(bptr + kk);
        acc = WMMA_F16(a, b, acc);
    }
    const int co0 = coTile + khalf * 8;
    float* yp = Y + ((size_t)(img * NPOS + pos)) * COUT_TOT + co0;
#pragma unroll
    for (int r = 0; r < 8; ++r) yp[r] = acc[r] + bias[co0 + r];
}

// ---------------------------------------------------------------------------
// Small prep kernels
// ---------------------------------------------------------------------------
__global__ void lss_zero_u32(unsigned int* p, int n) {
    int i = blockIdx.x * blockDim.x + threadIdx.x;
    if (i < n) p[i] = 0u;
}

// feat (NIMG, CI, FH, FW) f32 -> padded (NIMG, PPOS, CI) f16 interior
__global__ void lss_pack_feat(const float* __restrict__ in, _Float16* __restrict__ out, int n) {
    int tid = blockIdx.x * blockDim.x + threadIdx.x;
    if (tid >= n) return;
    int p  = tid % NPOS;
    int ci = (tid / NPOS) % CI;
    int i  = tid / (NPOS * CI);
    int h = p / FW, w = p % FW;
    out[((size_t)i * PPOS + (h + 1) * PFW + (w + 1)) * CI + ci] = (_Float16)in[tid];
}

// w (CO, CI, 3, 3) f32 -> (9, CO, CI) f16
__global__ void lss_pack_w3(const float* __restrict__ in, _Float16* __restrict__ out, int n) {
    int tid = blockIdx.x * blockDim.x + threadIdx.x;
    if (tid >= n) return;
    int t  = tid % 9;
    int ci = (tid / 9) % CI;
    int co = tid / (9 * CI);
    out[((size_t)t * CMID + co) * CI + ci] = (_Float16)in[tid];
}

// w_out (192, 512) f32 -> f16 same layout
__global__ void lss_pack_wout(const float* __restrict__ in, _Float16* __restrict__ out, int n) {
    int tid = blockIdx.x * blockDim.x + threadIdx.x;
    if (tid < n) out[tid] = (_Float16)in[tid];
}

// fold BN into scale/bias
__global__ void lss_bn_prep(const float* g, const float* b, const float* m,
                            const float* v, float* scale, float* bias, int n) {
    int i = blockIdx.x * blockDim.x + threadIdx.x;
    if (i >= n) return;
    float s = g[i] * rsqrtf(v[i] + 1e-5f);
    scale[i] = s;
    bias[i]  = b[i] - m[i] * s;
}

// in-place softmax over depth channels [0,112) of HF[pos][192]
__global__ void lss_softmax_depth(float* __restrict__ HF, int np) {
    int p = blockIdx.x * blockDim.x + threadIdx.x;
    if (p >= np) return;
    float* ptr = HF + (size_t)p * COUT_TOT;
    float mx = -3.4e38f;
    for (int d = 0; d < NDEPTH; ++d) mx = fmaxf(mx, ptr[d]);
    float s = 0.f;
    for (int d = 0; d < NDEPTH; ++d) { float e = expf(ptr[d] - mx); ptr[d] = e; s += e; }
    float inv = 1.f / s;
    for (int d = 0; d < NDEPTH; ++d) ptr[d] *= inv;
}

// ---------------------------------------------------------------------------
// 4x4 inverse (Gauss-Jordan w/ partial pivot) + matmul, per (b,n) camera.
// mats[bn*48 +  0..15] = inv(ida)
// mats[bn*48 + 16..31] = sensor2virtual @ inv(intrin)
// mats[bn*48 + 32..47] = bda @ sensor2ego @ inv(sensor2virtual)
// ---------------------------------------------------------------------------
__device__ void lss_inv4(const float* a_in, float* out) {
    float a[4][8];
    for (int r = 0; r < 4; ++r)
        for (int c = 0; c < 4; ++c) { a[r][c] = a_in[r * 4 + c]; a[r][4 + c] = (r == c) ? 1.f : 0.f; }
    for (int col = 0; col < 4; ++col) {
        int piv = col; float best = fabsf(a[col][col]);
        for (int r = col + 1; r < 4; ++r) {
            float v = fabsf(a[r][col]);
            if (v > best) { best = v; piv = r; }
        }
        if (piv != col)
            for (int c = 0; c < 8; ++c) { float t = a[col][c]; a[col][c] = a[piv][c]; a[piv][c] = t; }
        float inv = 1.f / a[col][col];
        for (int c = 0; c < 8; ++c) a[col][c] *= inv;
        for (int r = 0; r < 4; ++r) {
            if (r == col) continue;
            float f = a[r][col];
            for (int c = 0; c < 8; ++c) a[r][c] -= f * a[col][c];
        }
    }
    for (int r = 0; r < 4; ++r)
        for (int c = 0; c < 4; ++c) out[r * 4 + c] = a[r][4 + c];
}
__device__ void lss_mul4(const float* A, const float* Bm, float* C) {
    for (int r = 0; r < 4; ++r)
        for (int c = 0; c < 4; ++c) {
            float s = 0.f;
            for (int k = 0; k < 4; ++k) s += A[r * 4 + k] * Bm[k * 4 + c];
            C[r * 4 + c] = s;
        }
}
__global__ void lss_geom_prep(const float* __restrict__ s2e, const float* __restrict__ s2v,
                              const float* __restrict__ intrin, const float* __restrict__ ida,
                              const float* __restrict__ bda, float* __restrict__ mats) {
    int t = blockIdx.x * blockDim.x + threadIdx.x;
    if (t >= NIMG) return;
    int b = t / 6;
    float iida[16], ik[16], iv[16], cv[16], ce[16], tmp[16];
    lss_inv4(ida    + t * 16, iida);
    lss_inv4(intrin + t * 16, ik);
    lss_inv4(s2v    + t * 16, iv);
    lss_mul4(s2v + t * 16, ik, cv);
    lss_mul4(s2e + t * 16, iv, tmp);
    lss_mul4(bda + b * 16, tmp, ce);  // ce = bda @ s2e @ inv(s2v)
    float* o = mats + (size_t)t * 48;
    for (int i = 0; i < 16; ++i) { o[i] = iida[i]; o[16 + i] = cv[i]; o[32 + i] = ce[i]; }
}

// ---------------------------------------------------------------------------
// Geometry + voxel scatter. One block per (bn, pos); 128 threads.
// Phase 1: thread d<112 computes frustum point -> voxel sid (or -1), loads depth.
// Phase 2: 112x80 (d,c) pairs -> atomicAdd into out (B, 80, 128, 128).
// ---------------------------------------------------------------------------
__global__ __launch_bounds__(128) void lss_scatter_bev(
    const float* __restrict__ HF, const float* __restrict__ mats,
    const float* __restrict__ ref_heights, float* __restrict__ out)
{
    const int blk = blockIdx.x;          // 0 .. NIMG*NPOS-1
    const int p   = blk % NPOS;
    const int bn  = blk / NPOS;
    const int b   = bn / 6;
    const int w   = p % FW, h = p / FW;

    __shared__ float M[48];
    __shared__ float depthArr[NDEPTH];
    __shared__ float ctxArr[NCTX];
    __shared__ int   sidArr[NDEPTH];

    const int tid = threadIdx.x;
    if (tid < 48) M[tid] = mats[(size_t)bn * 48 + tid];
    const float* hfp = HF + ((size_t)bn * NPOS + p) * COUT_TOT;
    if (tid < NCTX) ctxArr[tid] = hfp[NDEPTH + tid];
    __syncthreads();

    if (tid < NDEPTH) {
        const float fx = w * (703.f / 43.f);
        const float fy = h * 17.f;                       // 255/15
        const float fd = 2.f + tid * (56.f / 111.f);
        // pts0 = inv(ida) @ [fx, fy, fd, 1]
        const float p0x = M[0]*fx + M[1]*fy + M[2]*fd  + M[3];
        const float p0y = M[4]*fx + M[5]*fy + M[6]*fd  + M[7];
        const float p0z = M[8]*fx + M[9]*fy + M[10]*fd + M[11];
        const float p0w = M[12]*fx+ M[13]*fy+ M[14]*fd + M[15];
        const float height = -p0z + ref_heights[bn];
        // pc = [10x, 10y, 10, w]
        const float cx = 10.f * p0x, cy = 10.f * p0y, cz = 10.f, cw = p0w;
        const float* C = M + 16;
        const float vx = C[0]*cx + C[1]*cy + C[2]*cz  + C[3]*cw;
        const float vy = C[4]*cx + C[5]*cy + C[6]*cz  + C[7]*cw;
        const float vz = C[8]*cx + C[9]*cy + C[10]*cz + C[11]*cw;
        const float ratio = height / vy;
        const float qx = vx * ratio, qy = vy * ratio, qz = vz * ratio;
        const float* T = M + 32;
        const float gx = T[0]*qx + T[1]*qy + T[2]*qz  + T[3];
        const float gy = T[4]*qx + T[5]*qy + T[6]*qz  + T[7];
        const float gz = T[8]*qx + T[9]*qy + T[10]*qz + T[11];
        // voxelize; C int-cast truncation matches .astype(int32)
        const int ix = (int)((gx + 51.2f) * 1.25f);
        const int iy = (int)((gy + 51.2f) * 1.25f);
        const int iz = (int)((gz + 5.0f)  * 0.125f);
        const bool valid = (ix >= 0) & (iy >= 0) & (iz >= 0) &
                           (ix < NX) & (iy < NY) & (iz < 1);
        sidArr[tid]   = valid ? (iy * NX + ix) : -1;
        depthArr[tid] = hfp[tid];
    }
    __syncthreads();

    float* ob = out + (size_t)b * NCTX * NY * NX;
    for (int pr = tid; pr < NDEPTH * NCTX; pr += 128) {
        const int d = pr / NCTX, c = pr % NCTX;
        const int sid = sidArr[d];
        if (sid >= 0) atomicAdd(ob + (size_t)c * (NY * NX) + sid, depthArr[d] * ctxArr[c]);
    }
}

// ---------------------------------------------------------------------------
// Host launcher
// ---------------------------------------------------------------------------
extern "C" void kernel_launch(void* const* d_in, const int* in_sizes, int n_in,
                              void* d_out, int out_size, void* d_ws, size_t ws_size,
                              hipStream_t stream) {
    const float* feat   = (const float*)d_in[0];
    const float* w1     = (const float*)d_in[1];
    const float* g1     = (const float*)d_in[2];
    const float* b1     = (const float*)d_in[3];
    const float* m1     = (const float*)d_in[4];
    const float* v1     = (const float*)d_in[5];
    const float* w2     = (const float*)d_in[6];
    const float* g2     = (const float*)d_in[7];
    const float* b2     = (const float*)d_in[8];
    const float* m2     = (const float*)d_in[9];
    const float* v2     = (const float*)d_in[10];
    const float* w_out  = (const float*)d_in[11];
    const float* b_out  = (const float*)d_in[12];
    const float* s2e    = (const float*)d_in[13];
    const float* s2v    = (const float*)d_in[14];
    const float* intrin = (const float*)d_in[15];
    const float* ida    = (const float*)d_in[16];
    const float* refh   = (const float*)d_in[17];
    const float* bda    = (const float*)d_in[18];
    float* out = (float*)d_out;

    // workspace carve-up (256B aligned)
    size_t off = 0;
    auto carve = [&](size_t bytes) -> char* {
        char* p = (char*)d_ws + off;
        off = (off + bytes + 255) & ~(size_t)255;
        return p;
    };
    const size_t padded_halves = (size_t)NIMG * PPOS * CI;      // per padded tensor
    _Float16* Fp  = (_Float16*)carve(padded_halves * 2);        // packed feat (padded)
    _Float16* H1  = (_Float16*)carve(padded_halves * 2);
    _Float16* H2  = (_Float16*)carve(padded_halves * 2);
    _Float16* W1p = (_Float16*)carve((size_t)9 * CMID * CI * 2);
    _Float16* W2p = (_Float16*)carve((size_t)9 * CMID * CI * 2);
    _Float16* Wop = (_Float16*)carve((size_t)COUT_TOT * CI * 2);
    float*    HF  = (float*)   carve((size_t)NIMG * NPOS * COUT_TOT * 4);
    float*    sc1 = (float*)   carve(CMID * 4);
    float*    bi1 = (float*)   carve(CMID * 4);
    float*    sc2 = (float*)   carve(CMID * 4);
    float*    bi2 = (float*)   carve(CMID * 4);
    float*    mats= (float*)   carve(NIMG * 48 * 4);
    (void)ws_size; (void)n_in; (void)in_sizes;

    // 0) zero output (atomics accumulate) + zero padded activation buffers
    lss_zero_u32<<<(out_size + 255) / 256, 256, 0, stream>>>((unsigned int*)out, out_size);
    {
        // Fp, H1, H2 are contiguous (each 256B-aligned size): one clear
        int n32 = (int)((padded_halves * 2 * 3 + 3) / 4);
        lss_zero_u32<<<(n32 + 255) / 256, 256, 0, stream>>>((unsigned int*)Fp, n32);
    }

    // 1) pack inputs to f16 WMMA-friendly layouts
    {
        int n = NIMG * CI * NPOS;
        lss_pack_feat<<<(n + 255) / 256, 256, 0, stream>>>(feat, Fp, n);
    }
    {
        int n = CMID * CI * 9;
        lss_pack_w3<<<(n + 255) / 256, 256, 0, stream>>>(w1, W1p, n);
        lss_pack_w3<<<(n + 255) / 256, 256, 0, stream>>>(w2, W2p, n);
    }
    {
        int n = COUT_TOT * CI;
        lss_pack_wout<<<(n + 255) / 256, 256, 0, stream>>>(w_out, Wop, n);
    }

    // 2) fold BN, prep geometry matrices
    lss_bn_prep<<<2, 256, 0, stream>>>(g1, b1, m1, v1, sc1, bi1, CMID);
    lss_bn_prep<<<2, 256, 0, stream>>>(g2, b2, m2, v2, sc2, bi2, CMID);
    lss_geom_prep<<<1, 32, 0, stream>>>(s2e, s2v, intrin, ida, bda, mats);

    // 3) conv stack on WMMA
    dim3 g3(NPOS / 64, CMID / 128, NIMG);       // (11, 4, 12)
    lss_conv3x3_wmma<<<g3, 256, 0, stream>>>(Fp, W1p, sc1, bi1, H1);
    lss_conv3x3_wmma<<<g3, 256, 0, stream>>>(H1, W2p, sc2, bi2, H2);
    dim3 g1x1(NPOS / 32, COUT_TOT / 64, NIMG);  // (22, 3, 12)
    lss_conv1x1_wmma<<<g1x1, 256, 0, stream>>>(H2, Wop, b_out, HF);

    // 4) depth softmax (in place over channels 0..111)
    {
        int np = NIMG * NPOS;
        lss_softmax_depth<<<(np + 127) / 128, 128, 0, stream>>>(HF, np);
    }

    // 5) lift-splat scatter into BEV
    lss_scatter_bev<<<NIMG * NPOS, 128, 0, stream>>>(HF, mats, refh, out);
}
